// Attention_4312147165344
// MI455X (gfx1250) — compile-verified, matrix operand order checked by reading, not run on hardware
//
#include <hip/hip_runtime.h>
#include <hip/hip_bf16.h>

// ---------------------------------------------------------------------------
// CDNA5 (gfx1250, wave32) fused attention block:
//   RMSNorm -> QKV GEMM -> flash attention (softcap+causal) -> out GEMM
// All matmuls via v_wmma_f32_16x16x32_f16 (fp32 accumulate).
// B/V operands stored in LDS in fragment-major (swizzled) order so every
// WMMA operand load is a pair of ds_load_b128 (no v_mov_b16 repacking).
// Tile fills are unrolled b128 copies (no serialized load->wait->store loops).
// Causal mask applied analytically (reference mask is tril), mask input unused.
// ---------------------------------------------------------------------------

typedef __attribute__((ext_vector_type(16))) _Float16 v16h;
typedef __attribute__((ext_vector_type(8)))  float    v8f;

union HFrag {
    v16h v;
    _Float16 h[16];
    uint4 q[2];
};

__device__ inline v8f wmma16(const HFrag& a, const HFrag& b, v8f c) {
    return __builtin_amdgcn_wmma_f32_16x16x32_f16(
        false, a.v, false, b.v, (short)0, c, false, false);
}

#define N_SEQ   2048
#define DIM     1024
#define HEADS   16
#define DHEAD   64
#define NQKV    3072
#define BATCH   2

// Fragment element <-> k mapping for 16-bit A/B operands (ISA 7.12.2):
//   kk(e, half) = e + (e>=8 ? 8 : 0) + 8*half
// Inverse (used when swizzling into LDS):
//   half(k) = (k>>3)&1 ;  e(k) = (k&7) | ((k>>4)<<3)

// ---------------------------------------------------------------------------
// Kernel 1: RMSNorm-variant + cast to f16.  xn = x/||x|| * sqrt(dim) * (g+1)
// ---------------------------------------------------------------------------
__global__ __launch_bounds__(256) void rmsnorm_kernel(
    const float* __restrict__ x, const float* __restrict__ gamma,
    _Float16* __restrict__ xn)
{
    __shared__ float red[256];
    const int row = blockIdx.x;
    const int tid = threadIdx.x;
    const float* xr = x + (size_t)row * DIM;
    float ss = 0.f;
#pragma unroll
    for (int it = 0; it < 4; ++it) {
        const float v = xr[tid + it * 256];
        ss += v * v;
    }
    red[tid] = ss;
    __syncthreads();
    for (int s = 128; s > 0; s >>= 1) {
        if (tid < s) red[tid] += red[tid + s];
        __syncthreads();
    }
    const float norm = sqrtf(red[0]);
    const float inv = 32.0f / fmaxf(norm, 1e-12f);   // sqrt(1024)=32
#pragma unroll
    for (int it = 0; it < 4; ++it) {
        const int c = tid + it * 256;
        xn[(size_t)row * DIM + c] = (_Float16)(xr[c] * inv * (gamma[c] + 1.0f));
    }
}

// ---------------------------------------------------------------------------
// Kernel 2: weight conversion fp32 -> fp16 (q-scale 0.125 folded into w_qkv).
// ---------------------------------------------------------------------------
__global__ __launch_bounds__(256) void convert_wqkv_kernel(
    const float* __restrict__ w, _Float16* __restrict__ o)
{
    const int idx = blockIdx.x * 256 + threadIdx.x;
    if (idx < DIM * NQKV) {
        const int col = idx % NQKV;
        const float s = (col < DIM) ? 0.125f : 1.0f;
        o[idx] = (_Float16)(w[idx] * s);
    }
}

__global__ __launch_bounds__(256) void convert_plain_kernel(
    const float* __restrict__ w, _Float16* __restrict__ o, int n)
{
    const int idx = blockIdx.x * 256 + threadIdx.x;
    if (idx < n) o[idx] = (_Float16)w[idx];
}

// ---------------------------------------------------------------------------
// Shared GEMM machinery: block 256 threads (8 waves as 2Mx4N),
// block tile 128x256, K-step 32, wave tile 64x64 -> 4x4 WMMA frags.
// sA row-major; sB fragment-major swizzled: sBs[nChunk][lane][elem].
// ---------------------------------------------------------------------------
#define GT_M 128
#define GT_N 256
#define GT_K 32

__device__ __forceinline__ void gemm_fill_tiles(
    const _Float16* __restrict__ A, const _Float16* __restrict__ B,
    int lda, int ldb, int m0, int n0, int k0, int tid,
    _Float16 (*sA)[GT_K], _Float16 (*sBs)[32][16])
{
    // A tile: 128x32 halfs = 512 uint4, row-major, 2 b128 copies per thread.
#pragma unroll
    for (int it = 0; it < 2; ++it) {
        const int idx = tid + it * 256;            // 0..511
        const int row = idx >> 2, c = idx & 3;
        ((uint4*)&sA[0][0])[idx] =
            *((const uint4*)(A + (size_t)(m0 + row) * lda + k0) + c);
    }
    // B tile: 32x256, swizzled into fragment-major order.
    // Each thread owns one column pair n2 and 8 k-quads (stride 4 rows).
    const int n2 = (tid & 127) << 1;               // even, fixed per thread
    const int k2b = (tid >> 7) << 1;               // 0 or 2
    const _Float16* Bp = B + (size_t)(k0 + k2b) * ldb + n0 + n2;
    const int f = n2 >> 4, rr = n2 & 15;
    unsigned int av[8], bv[8];
#pragma unroll
    for (int it = 0; it < 8; ++it) {               // issue all loads first
        av[it] = *(const unsigned int*)(Bp + (size_t)(4 * it) * ldb);
        bv[it] = *(const unsigned int*)(Bp + (size_t)(4 * it + 1) * ldb);
    }
#pragma unroll
    for (int it = 0; it < 8; ++it) {
        const int k2 = k2b + 4 * it;
        const unsigned int u0 = (av[it] & 0xffffu) | (bv[it] << 16);
        const unsigned int u1 = (av[it] >> 16) | (bv[it] & 0xffff0000u);
        const int halfk = (k2 >> 3) & 1;
        const int e = (k2 & 7) | ((k2 >> 4) << 3);
        ((unsigned int*)&sBs[f][halfk * 16 + rr][0])[e >> 1] = u0;
        ((unsigned int*)&sBs[f][halfk * 16 + rr + 1][0])[e >> 1] = u1;
    }
}

__device__ __forceinline__ void gemm_core(
    const _Float16 (*sA)[GT_K], const _Float16 (*sBs)[32][16],
    int waveM, int waveN, int lane, v8f acc[4][4])
{
    const int r    = lane & 15;
    const int half = lane >> 4;
    HFrag af[4], bfr[4];
#pragma unroll
    for (int fm = 0; fm < 4; ++fm) {
        const int row = waveM * 64 + fm * 16 + r;
        af[fm].q[0] = *(const uint4*)&sA[row][8 * half];
        af[fm].q[1] = *(const uint4*)&sA[row][16 + 8 * half];
    }
#pragma unroll
    for (int fn = 0; fn < 4; ++fn) {
        bfr[fn].v = *(const v16h*)&sBs[waveN * 4 + fn][lane][0];
    }
#pragma unroll
    for (int fm = 0; fm < 4; ++fm)
#pragma unroll
        for (int fn = 0; fn < 4; ++fn)
            acc[fm][fn] = wmma16(af[fm], bfr[fn], acc[fm][fn]);
}

// ---------------------------------------------------------------------------
// Kernel 3: QKV GEMM. C[4096,3072] = A[4096,1024]*B[1024,3072], scattered
// into q/k/v tensors of shape (b, h, n, d) f16.
// ---------------------------------------------------------------------------
__global__ __launch_bounds__(256) void gemm_qkv_kernel(
    const _Float16* __restrict__ A, const _Float16* __restrict__ B,
    _Float16* __restrict__ qb, _Float16* __restrict__ kb,
    _Float16* __restrict__ vb)
{
    __shared__ __attribute__((aligned(32))) _Float16 sA[GT_M][GT_K];
    __shared__ __attribute__((aligned(32))) _Float16 sBs[16][32][16];
    const int tid   = threadIdx.x;
    const int lane  = tid & 31;
    const int w     = tid >> 5;
    const int waveM = w & 1;
    const int waveN = w >> 1;
    const int m0 = blockIdx.y * GT_M;
    const int n0 = blockIdx.x * GT_N;

    v8f acc[4][4];
#pragma unroll
    for (int fm = 0; fm < 4; ++fm)
#pragma unroll
        for (int fn = 0; fn < 4; ++fn)
#pragma unroll
            for (int e = 0; e < 8; ++e) acc[fm][fn][e] = 0.f;

    for (int k0 = 0; k0 < DIM; k0 += GT_K) {
        gemm_fill_tiles(A, B, DIM, NQKV, m0, n0, k0, tid, sA, sBs);
        if (k0 + GT_K < DIM) {   // hint next tile into cache (global_prefetch_b8)
            __builtin_prefetch(A + (size_t)(m0 + (tid >> 1)) * DIM + k0 + GT_K, 0, 1);
            __builtin_prefetch(B + (size_t)(k0 + GT_K + (tid >> 3)) * NQKV + n0, 0, 1);
        }
        __syncthreads();
        gemm_core(sA, sBs, waveM, waveN, lane, acc);
        __syncthreads();
    }

    // Scatter epilogue: per-fn base pointer, stores use immediate offsets.
    const int r = lane & 15, half = lane >> 4;
    const int batch = m0 >> 11;
    const int iBase = (m0 & 2047) + waveM * 64 + 8 * half;
#pragma unroll
    for (int fn = 0; fn < 4; ++fn) {
        const int e = n0 + waveN * 64 + fn * 16 + r;
        const int which = e >> 10;
        const int hh    = (e >> 6) & 15;
        const int d     = e & 63;
        _Float16* base = (which == 0) ? qb : (which == 1) ? kb : vb;
        _Float16* dst = base +
            (((size_t)(batch * HEADS + hh)) * N_SEQ + iBase) * DHEAD + d;
#pragma unroll
        for (int fm = 0; fm < 4; ++fm)
#pragma unroll
            for (int vr = 0; vr < 8; ++vr)
                dst[(size_t)(fm * 16 + vr) * DHEAD] = (_Float16)acc[fm][fn][vr];
    }
}

// ---------------------------------------------------------------------------
// Kernel 4: flash attention per (b,h).  Block = 256 threads (8 waves),
// 128 query rows per block (16 per wave), K/V tiles of 64 keys.
// K row-major in LDS (A-like fragment reads contiguous);
// V fragment-major swizzled so PV B-frags are contiguous b128 pairs.
// ---------------------------------------------------------------------------
__global__ __launch_bounds__(256) void flash_kernel(
    const _Float16* __restrict__ q, const _Float16* __restrict__ k,
    const _Float16* __restrict__ v, _Float16* __restrict__ o)
{
    __shared__ __attribute__((aligned(32))) _Float16 sK[64][64];
    __shared__ __attribute__((aligned(32))) _Float16 sVs[4][2][32][16];
    __shared__ __attribute__((aligned(32))) _Float16 sP[8][16][64];

    const int tid  = threadIdx.x;
    const int lane = tid & 31;
    const int w    = tid >> 5;            // 0..7
    const int r    = lane & 15;
    const int half = lane >> 4;
    const int qt = blockIdx.x;            // 128-row query tile
    const int bh = blockIdx.y;
    const int qi0 = qt * 128;

    const _Float16* Qb = q + (size_t)bh * N_SEQ * DHEAD;
    const _Float16* Kb = k + (size_t)bh * N_SEQ * DHEAD;
    const _Float16* Vb = v + (size_t)bh * N_SEQ * DHEAD;

    // Q fragments (16 rows x 64 d): 2 A-frags, loaded as b128 quads.
    HFrag qa[2];
    const int qrow = qi0 + w * 16 + r;
#pragma unroll
    for (int c = 0; c < 2; ++c) {
        qa[c].q[0] = *(const uint4*)&Qb[(size_t)qrow * DHEAD + c * 32 + 8 * half];
        qa[c].q[1] = *(const uint4*)&Qb[(size_t)qrow * DHEAD + c * 32 + 16 + 8 * half];
    }

    float mrow[8], lrow[8];
    v8f of[4];
#pragma unroll
    for (int vr = 0; vr < 8; ++vr) { mrow[vr] = -3.0e38f; lrow[vr] = 0.f; }
#pragma unroll
    for (int nb = 0; nb < 4; ++nb)
#pragma unroll
        for (int e = 0; e < 8; ++e) of[nb][e] = 0.f;

    const int jTiles = 2 * qt + 2;        // keys up to qi0+127 (diagonal)
    for (int jt = 0; jt < jTiles; ++jt) {
        const int j0 = jt * 64;
        // K tile: row-major b128 copies (64x64 halfs = 512 uint4)
        {
            const uint4* gk = (const uint4*)(Kb + (size_t)j0 * DHEAD);
            uint4* lk = (uint4*)&sK[0][0];
#pragma unroll
            for (int it = 0; it < 2; ++it) lk[tid + it * 256] = gk[tid + it * 256];
        }
        // V tile: swizzle into fragment-major order via 2x2 quads.
        {
            const int d2 = (tid & 31) << 1;           // fixed per thread
            const int key2b = (tid >> 5) << 1;        // 0..14 even
            const _Float16* Vp = Vb + (size_t)(j0 + key2b) * DHEAD + d2;
            const int fd = d2 >> 4, rr = d2 & 15;
            unsigned int av[4], bv[4];
#pragma unroll
            for (int it = 0; it < 4; ++it) {          // issue all loads first
                av[it] = *(const unsigned int*)(Vp + (size_t)(16 * it) * DHEAD);
                bv[it] = *(const unsigned int*)(Vp + (size_t)(16 * it + 1) * DHEAD);
            }
#pragma unroll
            for (int it = 0; it < 4; ++it) {
                const int key2 = key2b + 16 * it;
                const unsigned int u0 = (av[it] & 0xffffu) | (bv[it] << 16);
                const unsigned int u1 = (av[it] >> 16) | (bv[it] & 0xffff0000u);
                const int c  = key2 >> 5;
                const int kc = key2 & 31;
                const int halfk = (kc >> 3) & 1;
                const int e = (kc & 7) | ((kc >> 4) << 3);
                ((unsigned int*)&sVs[fd][c][halfk * 16 + rr][0])[e >> 1] = u0;
                ((unsigned int*)&sVs[fd][c][halfk * 16 + rr + 1][0])[e >> 1] = u1;
            }
        }
        if (jt + 1 < jTiles) {
            __builtin_prefetch(Kb + (size_t)(j0 + 64 + (tid >> 2)) * DHEAD, 0, 1);
            __builtin_prefetch(Vb + (size_t)(j0 + 64 + (tid >> 2)) * DHEAD, 0, 1);
        }
        __syncthreads();

        // S = Q K^T over 4 key chunks of 16
        float sv_[4][8];
#pragma unroll
        for (int nb = 0; nb < 4; ++nb) {
            HFrag kf[2];
#pragma unroll
            for (int c = 0; c < 2; ++c) {
                kf[c].q[0] = *(const uint4*)&sK[nb * 16 + r][c * 32 + 8 * half];
                kf[c].q[1] = *(const uint4*)&sK[nb * 16 + r][c * 32 + 16 + 8 * half];
            }
            v8f s;
#pragma unroll
            for (int e = 0; e < 8; ++e) s[e] = 0.f;
            s = wmma16(qa[0], kf[0], s);
            s = wmma16(qa[1], kf[1], s);
#pragma unroll
            for (int vr = 0; vr < 8; ++vr) {
                float xv = s[vr];
                xv = 50.0f * tanhf(xv * 0.02f);
                const int ii = qi0 + w * 16 + vr + 8 * half;
                const int jj = j0 + nb * 16 + r;
                if (jj > ii) xv = -3.0e38f;
                sv_[nb][vr] = xv;
            }
        }

        // Online softmax (row reductions within the 16-lane groups)
        float alpha[8];
#pragma unroll
        for (int vr = 0; vr < 8; ++vr) {
            float mx = sv_[0][vr];
#pragma unroll
            for (int nb = 1; nb < 4; ++nb) mx = fmaxf(mx, sv_[nb][vr]);
            mx = fmaxf(mx, __shfl_xor(mx, 8, 32));
            mx = fmaxf(mx, __shfl_xor(mx, 4, 32));
            mx = fmaxf(mx, __shfl_xor(mx, 2, 32));
            mx = fmaxf(mx, __shfl_xor(mx, 1, 32));
            const float mnew = fmaxf(mrow[vr], mx);
            alpha[vr] = __expf(mrow[vr] - mnew);
            mrow[vr] = mnew;
            float psum = 0.f;
#pragma unroll
            for (int nb = 0; nb < 4; ++nb) {
                const float p = __expf(sv_[nb][vr] - mnew);
                sv_[nb][vr] = p;
                psum += p;
            }
            psum += __shfl_xor(psum, 8, 32);
            psum += __shfl_xor(psum, 4, 32);
            psum += __shfl_xor(psum, 2, 32);
            psum += __shfl_xor(psum, 1, 32);
            lrow[vr] = lrow[vr] * alpha[vr] + psum;
        }

        // Stage P (wave-private C-layout -> LDS -> A-layout)
#pragma unroll
        for (int nb = 0; nb < 4; ++nb)
#pragma unroll
            for (int vr = 0; vr < 8; ++vr)
                sP[w][vr + 8 * half][nb * 16 + r] = (_Float16)sv_[nb][vr];

        HFrag pa[2];
#pragma unroll
        for (int c = 0; c < 2; ++c) {
            pa[c].q[0] = *(const uint4*)&sP[w][r][c * 32 + 8 * half];
            pa[c].q[1] = *(const uint4*)&sP[w][r][c * 32 + 16 + 8 * half];
        }

        // Rescale accumulators, then O += P V
#pragma unroll
        for (int nb = 0; nb < 4; ++nb)
#pragma unroll
            for (int vr = 0; vr < 8; ++vr) of[nb][vr] *= alpha[vr];

#pragma unroll
        for (int nb = 0; nb < 4; ++nb) {
            HFrag vf0, vf1;
            vf0.v = *(const v16h*)&sVs[nb][0][lane][0];
            vf1.v = *(const v16h*)&sVs[nb][1][lane][0];
            of[nb] = wmma16(pa[0], vf0, of[nb]);
            of[nb] = wmma16(pa[1], vf1, of[nb]);
        }
        __syncthreads();
    }

    // Normalize and store to (b, n, h*d) f16 layout for the out projection.
    const int batch = bh >> 4;
    const int hh    = bh & 15;
#pragma unroll
    for (int nb = 0; nb < 4; ++nb) {
        _Float16* dst = o +
            ((size_t)batch * N_SEQ + qi0 + w * 16 + 8 * half) * DIM +
            hh * DHEAD + nb * 16 + r;
#pragma unroll
        for (int vr = 0; vr < 8; ++vr)
            dst[(size_t)vr * DIM] = (_Float16)(of[nb][vr] / lrow[vr]);
    }
}

// ---------------------------------------------------------------------------
// Kernel 5: output projection. out[4096,1024] fp32 = A[4096,1024]*B[1024,1024]
// ---------------------------------------------------------------------------
__global__ __launch_bounds__(256) void gemm_out_kernel(
    const _Float16* __restrict__ A, const _Float16* __restrict__ B,
    float* __restrict__ out)
{
    __shared__ __attribute__((aligned(32))) _Float16 sA[GT_M][GT_K];
    __shared__ __attribute__((aligned(32))) _Float16 sBs[16][32][16];
    const int tid   = threadIdx.x;
    const int lane  = tid & 31;
    const int w     = tid >> 5;
    const int waveM = w & 1;
    const int waveN = w >> 1;
    const int m0 = blockIdx.y * GT_M;
    const int n0 = blockIdx.x * GT_N;

    v8f acc[4][4];
#pragma unroll
    for (int fm = 0; fm < 4; ++fm)
#pragma unroll
        for (int fn = 0; fn < 4; ++fn)
#pragma unroll
            for (int e = 0; e < 8; ++e) acc[fm][fn][e] = 0.f;

    for (int k0 = 0; k0 < DIM; k0 += GT_K) {
        gemm_fill_tiles(A, B, DIM, DIM, m0, n0, k0, tid, sA, sBs);
        if (k0 + GT_K < DIM) {
            __builtin_prefetch(A + (size_t)(m0 + (tid >> 1)) * DIM + k0 + GT_K, 0, 1);
            __builtin_prefetch(B + (size_t)(k0 + GT_K + (tid >> 3)) * DIM + n0, 0, 1);
        }
        __syncthreads();
        gemm_core(sA, sBs, waveM, waveN, lane, acc);
        __syncthreads();
    }

    const int r = lane & 15, half = lane >> 4;
#pragma unroll
    for (int fn = 0; fn < 4; ++fn) {
        const int e = n0 + waveN * 64 + fn * 16 + r;
        float* dst = out + (size_t)(m0 + waveM * 64 + 8 * half) * DIM + e;
#pragma unroll
        for (int fm = 0; fm < 4; ++fm)
#pragma unroll
            for (int vr = 0; vr < 8; ++vr)
                dst[(size_t)(fm * 16 + vr) * DIM] = acc[fm][fn][vr];
    }
}

// ---------------------------------------------------------------------------
// Host launcher.  Inputs: x, attn_mask(unused), gamma, w_qkv, w_out.
// Workspace layout (bytes):
//   [0,8M)    xn f16 (4096x1024)
//   [8M,14M)  w_qkv f16 (q-scale folded)
//   [14M,16M) w_out f16
//   [16M,24M) q (b,h,n,d) f16   [24M,32M) k   [32M,40M) v
//   [40M,48M) attn out (b,n,h*d) f16
// ---------------------------------------------------------------------------
extern "C" void kernel_launch(void* const* d_in, const int* in_sizes, int n_in,
                              void* d_out, int out_size, void* d_ws, size_t ws_size,
                              hipStream_t stream) {
    const float* x     = (const float*)d_in[0];
    // d_in[1]: attn_mask (tril) — causal mask applied analytically, not read.
    const float* gamma = (const float*)d_in[2];
    const float* w_qkv = (const float*)d_in[3];
    const float* w_out = (const float*)d_in[4];
    float* out = (float*)d_out;

    char* ws = (char*)d_ws;
    const size_t MB = 1024 * 1024;
    _Float16* xn   = (_Float16*)(ws);
    _Float16* wqh  = (_Float16*)(ws + 8 * MB);
    _Float16* woh  = (_Float16*)(ws + 14 * MB);
    _Float16* qb   = (_Float16*)(ws + 16 * MB);
    _Float16* kb   = (_Float16*)(ws + 24 * MB);
    _Float16* vb   = (_Float16*)(ws + 32 * MB);
    _Float16* ab   = (_Float16*)(ws + 40 * MB);

    rmsnorm_kernel<<<BATCH * N_SEQ, 256, 0, stream>>>(x, gamma, xn);
    convert_wqkv_kernel<<<(DIM * NQKV) / 256, 256, 0, stream>>>(w_qkv, wqh);
    convert_plain_kernel<<<(DIM * DIM) / 256, 256, 0, stream>>>(w_out, woh, DIM * DIM);
    gemm_qkv_kernel<<<dim3(NQKV / GT_N, (BATCH * N_SEQ) / GT_M), 256, 0, stream>>>(
        xn, wqh, qb, kb, vb);
    flash_kernel<<<dim3(N_SEQ / 128, BATCH * HEADS), 256, 0, stream>>>(qb, kb, vb, ab);
    gemm_out_kernel<<<dim3(DIM / GT_N, (BATCH * N_SEQ) / GT_M), 256, 0, stream>>>(
        ab, woh, out);
}
